// GptOssTopKRouter_90778428768672
// MI455X (gfx1250) — matrix operand order repaired and checked by reading, not compiled, not run on hardware
//
#include <hip/hip_runtime.h>

typedef __attribute__((ext_vector_type(16))) _Float16 v16h;
typedef __attribute__((ext_vector_type(8)))  _Float16 v8h;
typedef __attribute__((ext_vector_type(8)))  float    v8f;

#define T_TOK   16384
#define H_DIM   2880
#define E_NUM   128
#define KSTEP   32
#define KITERS  (H_DIM / KSTEP)      // 90
#define ROWS_WG 128
#define LPAD    132                  // padded logits row stride (floats)
#define BKPAD   40                   // halves per expert column (32 used + 8 pad)
#define BBUF_H  (E_NUM * BKPAD)      // halves per staging buffer

// Scheduling-intent directives (DS-read / WMMA interleave control).
// Mask bits: 0x008 = MFMA/WMMA, 0x100 = DS read.
#if defined(__has_builtin)
#if __has_builtin(__builtin_amdgcn_sched_group_barrier)
#define SCHED_GROUP(mask, size, id) __builtin_amdgcn_sched_group_barrier((mask), (size), (id))
#endif
#endif
#ifndef SCHED_GROUP
#define SCHED_GROUP(mask, size, id)
#endif

__device__ __forceinline__ v16h load_bfrag(const _Float16* bbase, int nt)
{
    const _Float16* bp = bbase + nt * 16 * BKPAD;
    v8h b0 = *(const v8h*)(bp);
    v8h b1 = *(const v8h*)(bp + 16);
    return __builtin_shufflevector(b0, b1,
        0, 1, 2, 3, 4, 5, 6, 7, 8, 9, 10, 11, 12, 13, 14, 15);
}

__global__ __launch_bounds__(256)
void router_topk_kernel(const float* __restrict__ hs,
                        const float* __restrict__ wt,
                        const float* __restrict__ bias,
                        float* __restrict__ out)
{
    // Phase 1: smem = double-buffered f16 weight chunk (2 * 128 * 40 * 2 = 20480 B)
    // Phase 2: smem = 64-row logits buffer (64 * 132 * 4 = 33792 B)
    __shared__ __align__(16) unsigned char smem[(ROWS_WG / 2) * LPAD * 4];
    __shared__ float bias_s[E_NUM];

    _Float16* Blds = (_Float16*)smem;
    float*    logb = (float*)smem;

    const int t    = threadIdx.x;       // 0..255
    const int wave = t >> 5;            // 0..7
    const int lane = t & 31;
    const int l15  = lane & 15;
    const int hi   = lane >> 4;         // 0 or 1 (half-wave)

    // weight-staging role: thread t handles expert column `se`, k-rows skb..skb+15
    const int se  = t & (E_NUM - 1);
    const int skb = (t >> 7) * 16;

    if (t < E_NUM) bias_s[t] = bias[t];

    // Stage k-chunk 0 into buffer 0 (fp32 -> f16, column-major per expert,
    // k contiguous per column -> two 16B LDS stores)
    {
        const float* wp = wt + (size_t)skb * E_NUM + se;
        _Float16* dst = Blds + se * BKPAD + skb;
        v8h w0, w1;
#pragma unroll
        for (int i = 0; i < 8; ++i) w0[i] = (_Float16)wp[(size_t)i * E_NUM];
#pragma unroll
        for (int i = 0; i < 8; ++i) w1[i] = (_Float16)wp[(size_t)(i + 8) * E_NUM];
        *(v8h*)(dst)     = w0;
        *(v8h*)(dst + 8) = w1;
    }

    // Each lane walks one row of hs (wave owns a 16-row M tile)
    const float* arow =
        hs + (size_t)(blockIdx.x * ROWS_WG + wave * 16 + l15) * H_DIM;

    v8f acc[8];
#pragma unroll
    for (int nt = 0; nt < 8; ++nt)
#pragma unroll
        for (int r = 0; r < 8; ++r) acc[nt][r] = 0.0f;

    for (int k = 0; k < KITERS; ++k) {
        __syncthreads();                       // buf (k&1) fully staged
        const int buf = k & 1;
        const int kc  = k * KSTEP;

        // --- A global loads first: get HBM traffic in flight ASAP.
        // 16-bit A 16x32 layout: lanes 0-15 K 0..7,16..23 ; lanes 16-31 K 8..15,24..31
        const float* ap = arow + kc + hi * 8;
        float4 a0 = *(const float4*)(ap);
        float4 a1 = *(const float4*)(ap + 4);
        float4 a2 = *(const float4*)(ap + 16);
        float4 a3 = *(const float4*)(ap + 20);

        // B fragment base for this k-chunk.
        // 16-bit B 32x16 layout: lanes 0-15 col=l15 K 0..7,16..23 ; lanes 16-31 K 8..15,24..31
        const _Float16* bbase = Blds + buf * BBUF_H + l15 * BKPAD + hi * 8;

        // --- Prime a depth-3 rotating window of B fragments (24 VGPRs live).
        v16h Bf0 = load_bfrag(bbase, 0);
        v16h Bf1 = load_bfrag(bbase, 1);
        v16h Bf2 = load_bfrag(bbase, 2);

        // --- Stage next weight chunk into the other buffer (safe: barrier
        // above proved everyone finished reading it in iteration k-1)
        if (k + 1 < KITERS) {
            const int kc1 = (k + 1) * KSTEP;
            const float* wp = wt + (size_t)(kc1 + skb) * E_NUM + se;
            _Float16* dst = Blds + (buf ^ 1) * BBUF_H + se * BKPAD + skb;
            v8h w0, w1;
#pragma unroll
            for (int i = 0; i < 8; ++i) w0[i] = (_Float16)wp[(size_t)i * E_NUM];
#pragma unroll
            for (int i = 0; i < 8; ++i) w1[i] = (_Float16)wp[(size_t)(i + 8) * E_NUM];
            *(v8h*)(dst)     = w0;
            *(v8h*)(dst + 8) = w1;
        }
        if (k + 2 < KITERS) {
            __builtin_prefetch(wt + (size_t)((k + 2) * KSTEP + skb) * E_NUM + se, 0, 1);
        }

        // --- Convert A to f16 fragment
        v16h A;
        A[0]  = (_Float16)a0.x; A[1]  = (_Float16)a0.y;
        A[2]  = (_Float16)a0.z; A[3]  = (_Float16)a0.w;
        A[4]  = (_Float16)a1.x; A[5]  = (_Float16)a1.y;
        A[6]  = (_Float16)a1.z; A[7]  = (_Float16)a1.w;
        A[8]  = (_Float16)a2.x; A[9]  = (_Float16)a2.y;
        A[10] = (_Float16)a2.z; A[11] = (_Float16)a2.w;
        A[12] = (_Float16)a3.x; A[13] = (_Float16)a3.y;
        A[14] = (_Float16)a3.z; A[15] = (_Float16)a3.w;

        // --- 8 WMMAs, software-pipelined against the B-fragment loads.
#pragma unroll
        for (int nt = 0; nt < 8; ++nt) {
            v16h Bnext;
            if (nt + 3 < 8) Bnext = load_bfrag(bbase, nt + 3);
            acc[nt] = __builtin_amdgcn_wmma_f32_16x16x32_f16(
                false, A, false, Bf0, (short)0, acc[nt], false, false);
            Bf0 = Bf1;
            Bf1 = Bf2;
            Bf2 = Bnext;
        }

        // --- Scheduling contract for this iteration's body:
        //   prime 3 fragments (6 DS reads), then 5x {1 WMMA, 2 DS reads},
        //   then the 3 drain WMMAs. Every WMMA retires with >=2 fragment
        //   loads still in flight (dscnt waits of ~4 instead of full drains).
        SCHED_GROUP(0x100, 6, 0);          // 6 DS reads (fragments 0..2)
#pragma unroll
        for (int s = 0; s < 5; ++s) {
            SCHED_GROUP(0x008, 1, 0);      // WMMA nt=s
            SCHED_GROUP(0x100, 2, 0);      // DS reads for fragment s+3
        }
        SCHED_GROUP(0x008, 3, 0);          // WMMA nt=5..7
    }

    __syncthreads();   // GEMM done; smem re-purposed as 64-row logits buffer

    const size_t out_row_base = (size_t)blockIdx.x * ROWS_WG;

#pragma unroll 1
    for (int phase = 0; phase < 2; ++phase) {
        // Waves of this phase dump their 16x128 tile (+bias) to LDS.
        if ((wave >> 2) == phase) {
            const int wrow0 = (wave & 3) * 16 + hi * 8;   // C layout: lanes<16 -> M=r, lanes>=16 -> M=8+r
#pragma unroll
            for (int nt = 0; nt < 8; ++nt) {
                const int col = nt * 16 + l15;
                const float bv = bias_s[col];
#pragma unroll
                for (int r = 0; r < 8; ++r)
                    logb[(wrow0 + r) * LPAD + col] = acc[nt][r] + bv;
            }
        }
        __syncthreads();

        // 64 threads scan one row each: strict-> insertion top-4 (stable ties,
        // matches jax.lax.top_k descending + lowest-index-first), then softmax
        // over the 4 kept values.
        if (t < 64) {
            const float* lr = logb + t * LPAD;
            float v0 = -3.402823466e38f, v1 = v0, v2 = v0, v3 = v0;
            int   i0 = 0, i1 = 0, i2 = 0, i3 = 0;
#pragma unroll
            for (int e4 = 0; e4 < E_NUM; e4 += 4) {
                float4 q = *(const float4*)(lr + e4);
                float vv[4] = {q.x, q.y, q.z, q.w};
#pragma unroll
                for (int j = 0; j < 4; ++j) {
                    float v = vv[j]; int e = e4 + j;
                    if (v > v3) {
                        if (v > v2) {
                            if (v > v1) {
                                if (v > v0) { v3=v2;i3=i2; v2=v1;i2=i1; v1=v0;i1=i0; v0=v;i0=e; }
                                else        { v3=v2;i3=i2; v2=v1;i2=i1; v1=v; i1=e; }
                            } else          { v3=v2;i3=i2; v2=v; i2=e; }
                        } else              { v3=v; i3=e; }
                    }
                }
            }
            const float e1 = __expf(v1 - v0);
            const float e2 = __expf(v2 - v0);
            const float e3 = __expf(v3 - v0);
            const float inv = 1.0f / (1.0f + e1 + e2 + e3);

            const size_t grow = out_row_base + (size_t)phase * 64 + t;
            float* wout = out + grow * 4;
            wout[0] = inv; wout[1] = e1 * inv; wout[2] = e2 * inv; wout[3] = e3 * inv;
            int* iout = (int*)out + (size_t)T_TOK * 4 + grow * 4;
            iout[0] = i0; iout[1] = i1; iout[2] = i2; iout[3] = i3;
        }
        __syncthreads();   // protect logb before next phase overwrites it
    }
}

extern "C" void kernel_launch(void* const* d_in, const int* in_sizes, int n_in,
                              void* d_out, int out_size, void* d_ws, size_t ws_size,
                              hipStream_t stream) {
    const float* hs   = (const float*)d_in[0];   // [16384, 2880] fp32
    const float* wt   = (const float*)d_in[1];   // [2880, 128]  fp32
    const float* bias = (const float*)d_in[2];   // [128]        fp32
    // d_in[3] = top_k (==4), baked in as a compile-time constant
    (void)in_sizes; (void)n_in; (void)out_size; (void)d_ws; (void)ws_size;

    router_topk_kernel<<<T_TOK / ROWS_WG, 256, 0, stream>>>(hs, wt, bias, (float*)d_out);
}
